// SnnLayer_59777354826392
// MI455X (gfx1250) — matrix operand order repaired and checked by reading, not compiled, not run on hardware
//
#include <hip/hip_runtime.h>
#include <math.h>

typedef __attribute__((ext_vector_type(16))) _Float16 v16h;
typedef __attribute__((ext_vector_type(8)))  _Float16 v8h;
typedef __attribute__((ext_vector_type(4)))  _Float16 v4h;
typedef __attribute__((ext_vector_type(8)))  float    v8f;
typedef __attribute__((ext_vector_type(4)))  float    v4f;

#define BATCH   1024
#define HID     4096
#define RR      2048
#define KSTEP   32
#define STEPS_PER_SEG (RR / KSTEP)    // 64
#define BM      128
#define BN      128
#define LDSW    40          // halves per LDS row: 32 data + 8 pad (80 B -> 16B-aligned chunks)
#define NTHREADS 512        // 16 waves: 4x4 wave grid, 32x32 wave tile -> low VGPR pressure

struct Seg { const float* Ap; const float* Wp; int lda; int aof; };

static __device__ __forceinline__ Seg seg_for(
    int step, bool r_in,
    const float* x_t, const float* spk_t,
    const float* W_x2in, const float* W_rec4in, const float* W_out2in,
    const float* W_rec4out, const float* W_in2out)
{
    const int seg = step >> 6;          // step / STEPS_PER_SEG
    Seg s;
    if (!r_in) {                        // r_out_: spk_out@W_rec4out + spk_in@W_in2out
        s.Ap  = spk_t; s.lda = HID;
        s.aof = (seg == 0) ? 0 : RR;
        s.Wp  = (seg == 0) ? W_rec4out : W_in2out;
    } else {                            // r_in_: x@W_x2in + spk_in@W_rec4in + spk_out@W_out2in
        if (seg == 0)      { s.Ap = x_t;   s.lda = RR;  s.aof = 0;  s.Wp = W_x2in;   }
        else if (seg == 1) { s.Ap = spk_t; s.lda = HID; s.aof = RR; s.Wp = W_rec4in; }
        else               { s.Ap = spk_t; s.lda = HID; s.aof = 0;  s.Wp = W_out2in; }
    }
    return s;
}

static __device__ __forceinline__ v4h cvt4(v4f v) {
    v4h h; h[0] = (_Float16)v[0]; h[1] = (_Float16)v[1];
           h[2] = (_Float16)v[2]; h[3] = (_Float16)v[3];
    return h;
}

__global__ __launch_bounds__(NTHREADS)
void snn_fused_wmma_kernel(
    const float* __restrict__ x_t,
    const float* __restrict__ mem_t,
    const float* __restrict__ spk_t,
    const float* __restrict__ b_t,
    const float* __restrict__ W_x2in,   const float* __restrict__ b_x2in,
    const float* __restrict__ W_rec4in, const float* __restrict__ b_rec4in,
    const float* __restrict__ W_in2out, const float* __restrict__ b_in2out,
    const float* __restrict__ W_rec4out,const float* __restrict__ b_rec4out,
    const float* __restrict__ W_out2in, const float* __restrict__ b_out2in,
    const float* __restrict__ tau_adp,
    const float* __restrict__ tau_m,
    float* __restrict__ out)
{
    __shared__ _Float16 sA[2][BM][LDSW];   // A tile: [m][k]  (f16), double buffered
    __shared__ _Float16 sB[2][BN][LDSW];   // W tile: [n][k]  (f16)  (B[k][n] = W[n][k])

    const int tid  = threadIdx.x;
    const int lane = tid & 31;
    const int wave = tid >> 5;          // 0..15
    const int wm   = wave & 3;          // 4 waves along M -> 4*32 = 128
    const int wn   = wave >> 2;         // 4 waves along N -> 4*32 = 128
    const int lidx = lane & 15;
    const int hsel = lane >> 4;         // 0 or 1 (half-wave select)

    const int m0    = blockIdx.y * BM;              // batch-row base
    const int nblk  = blockIdx.x;                   // 0..31
    const bool r_in = (nblk >= 16);                 // columns 2048..4095 -> r_in_
    const int nl0   = (nblk & 15) * BN;             // W row base within half
    const int nbase = nblk * BN;                    // global column base in HID

    const int nsteps = (r_in ? 3 : 2) * STEPS_PER_SEG;

    // cooperative-load coordinates: 512 threads cover a 128x32 f32 tile in 2 float4 passes
    const int colg = (tid & 7) << 2;    // 0,4,...,28
    const int row0 = tid >> 3;          // 0..63
    const int row1 = row0 + 64;

    v8f acc[2][2];
#pragma unroll
    for (int i = 0; i < 2; ++i)
#pragma unroll
        for (int j = 0; j < 2; ++j)
            acc[i][j] = (v8f)0.0f;

    // ---- staging registers (plain scalars: must stay in VGPRs) ----
    v4f rA0, rA1, rB0, rB1;

    // ---- prologue: stage + store tile 0 ----
    {
        Seg sg = seg_for(0, r_in, x_t, spk_t, W_x2in, W_rec4in, W_out2in, W_rec4out, W_in2out);
        rA0 = *(const v4f*)(sg.Ap + (size_t)(m0 + row0) * sg.lda + sg.aof + colg);
        rA1 = *(const v4f*)(sg.Ap + (size_t)(m0 + row1) * sg.lda + sg.aof + colg);
        rB0 = *(const v4f*)(sg.Wp + (size_t)(nl0 + row0) * RR + colg);
        rB1 = *(const v4f*)(sg.Wp + (size_t)(nl0 + row1) * RR + colg);
        *(v4h*)&sA[0][row0][colg] = cvt4(rA0);
        *(v4h*)&sA[0][row1][colg] = cvt4(rA1);
        *(v4h*)&sB[0][row0][colg] = cvt4(rB0);
        *(v4h*)&sB[0][row1][colg] = cvt4(rB1);
    }
    __syncthreads();

    int cur = 0;
#pragma unroll 1
    for (int step = 0; step < nsteps; ++step) {
        const bool has_next = (step + 1 < nsteps);

        if (has_next) {     // issue next-tile global loads; they stay in flight during WMMA
            Seg sg = seg_for(step + 1, r_in, x_t, spk_t, W_x2in, W_rec4in, W_out2in, W_rec4out, W_in2out);
            const int k0 = ((step + 1) & (STEPS_PER_SEG - 1)) * KSTEP;
            rA0 = *(const v4f*)(sg.Ap + (size_t)(m0 + row0) * sg.lda + sg.aof + k0 + colg);
            rA1 = *(const v4f*)(sg.Ap + (size_t)(m0 + row1) * sg.lda + sg.aof + k0 + colg);
            rB0 = *(const v4f*)(sg.Wp + (size_t)(nl0 + row0) * RR + k0 + colg);
            rB1 = *(const v4f*)(sg.Wp + (size_t)(nl0 + row1) * RR + k0 + colg);
        }

        // A fragments: 16-bit A 16x32 layout
        // lane<16: M=lidx, halves 0..7 = K0..7,  8..15 = K16..23
        // lane>=16: M=lidx, halves 0..7 = K8..15, 8..15 = K24..31
        v16h afrag[2];
#pragma unroll
        for (int i = 0; i < 2; ++i) {
            const _Float16* pa = &sA[cur][wm * 32 + i * 16 + lidx][hsel * 8];
            v8h lo = *(const v8h*)pa;
            v8h hi = *(const v8h*)(pa + 16);
#pragma unroll
            for (int t = 0; t < 8; ++t) { afrag[i][t] = lo[t]; afrag[i][8 + t] = hi[t]; }
        }
        // B fragments: 32x16 layout: N = lidx, halves 0..7 = K hsel*8+0..7, 8..15 = K 16+hsel*8+0..7
        v16h bfrag[2];
#pragma unroll
        for (int j = 0; j < 2; ++j) {
            const _Float16* pb = &sB[cur][wn * 32 + j * 16 + lidx][hsel * 8];
            v8h lo = *(const v8h*)pb;
            v8h hi = *(const v8h*)(pb + 16);
#pragma unroll
            for (int t = 0; t < 8; ++t) { bfrag[j][t] = lo[t]; bfrag[j][8 + t] = hi[t]; }
        }
#pragma unroll
        for (int i = 0; i < 2; ++i)
#pragma unroll
            for (int j = 0; j < 2; ++j)
                acc[i][j] = __builtin_amdgcn_wmma_f32_16x16x32_f16(
                    false, afrag[i], false, bfrag[j],
                    (short)0, acc[i][j], false, false);

        if (has_next) {
            const int nb = cur ^ 1;   // safe: nb last read at step-1, fenced by that barrier
            *(v4h*)&sA[nb][row0][colg] = cvt4(rA0);
            *(v4h*)&sA[nb][row1][colg] = cvt4(rA1);
            *(v4h*)&sB[nb][row0][colg] = cvt4(rB0);
            *(v4h*)&sB[nb][row1][colg] = cvt4(rB1);
            __syncthreads();
            cur = nb;
        }
    }

    // ---- fused SNN epilogue ----
    // C layout: VGPR r -> M = r + 8*hsel, N = lidx
    const size_t S = (size_t)BATCH * HID;
    const int mrow0 = m0 + wm * 32;

#pragma unroll
    for (int j = 0; j < 2; ++j) {
        const int ng = nbase + wn * 32 + j * 16 + lidx;       // global hidden column
        float bias;
        if (!r_in) bias = b_rec4out[ng] + b_in2out[ng];
        else {
            const int nl = ng - RR;
            bias = b_x2in[nl] + b_rec4in[nl] + b_out2in[nl];
        }
        const float ts_m = 1.0f / (1.0f + expf(-tau_m[ng]));
        const float ts_a = 1.0f / (1.0f + expf(-tau_adp[ng]));
#pragma unroll
        for (int i = 0; i < 2; ++i) {
#pragma unroll
            for (int r = 0; r < 8; ++r) {
                const int m = mrow0 + i * 16 + (hsel << 3) + r;
                const size_t idx = (size_t)m * HID + ng;
                const float spk  = spk_t[idx];
                const float bold = b_t[idx];
                const float memo = mem_t[idx];
                const float bnew = ts_a * bold + (1.0f - ts_a) * spk;
                const float thre = 0.1f + 1.8f * bnew;          // BASELINE_THRE + BETA*b
                const float inp  = acc[i][j][r] + bias;
                const float mem  = memo * ts_m + (1.0f - ts_m) * 3.0f * inp - thre * spk; // R_M=3, DT=1
                const float spike = (mem - thre) > 0.0f ? 1.0f : 0.0f;
                out[idx]         = mem;
                out[S + idx]     = spike;
                out[2 * S + idx] = bnew;
            }
        }
    }
}

extern "C" void kernel_launch(void* const* d_in, const int* in_sizes, int n_in,
                              void* d_out, int out_size, void* d_ws, size_t ws_size,
                              hipStream_t stream) {
    (void)in_sizes; (void)n_in; (void)d_ws; (void)ws_size; (void)out_size;
    const float* x_t       = (const float*)d_in[0];
    const float* mem_t     = (const float*)d_in[1];
    const float* spk_t     = (const float*)d_in[2];
    const float* b_t       = (const float*)d_in[3];
    const float* W_x2in    = (const float*)d_in[4];
    const float* b_x2in    = (const float*)d_in[5];
    const float* W_rec4in  = (const float*)d_in[6];
    const float* b_rec4in  = (const float*)d_in[7];
    const float* W_in2out  = (const float*)d_in[8];
    const float* b_in2out  = (const float*)d_in[9];
    const float* W_rec4out = (const float*)d_in[10];
    const float* b_rec4out = (const float*)d_in[11];
    const float* W_out2in  = (const float*)d_in[12];
    const float* b_out2in  = (const float*)d_in[13];
    const float* tau_adp   = (const float*)d_in[14];
    const float* tau_m     = (const float*)d_in[15];
    float* out = (float*)d_out;

    dim3 grid(HID / BN, BATCH / BM, 1);   // 32 x 8
    dim3 block(NTHREADS, 1, 1);
    snn_fused_wmma_kernel<<<grid, block, 0, stream>>>(
        x_t, mem_t, spk_t, b_t,
        W_x2in, b_x2in, W_rec4in, b_rec4in,
        W_in2out, b_in2out, W_rec4out, b_rec4out,
        W_out2in, b_out2in, tau_adp, tau_m, out);
}
